// DeepRNN_3977139716152
// MI455X (gfx1250) — matrix operand order repaired and checked by reading, not compiled, not run on hardware
//
#include <hip/hip_runtime.h>
#include <cstdint>
#include <cstddef>

// ---------------------------------------------------------------------------
// Problem constants (from reference)
// ---------------------------------------------------------------------------
#define VOCAB  32000
#define EMBED  512
#define HIDDEN 1024
#define BATCH  16
#define TLEN   256

// ---------------------------------------------------------------------------
// Types
// ---------------------------------------------------------------------------
typedef __attribute__((ext_vector_type(16))) __bf16 v16bf;
typedef __attribute__((ext_vector_type(8)))  __bf16 v8bf;
typedef __attribute__((ext_vector_type(4)))  __bf16 v4bf;
typedef __attribute__((ext_vector_type(8)))  float  v8f;

__device__ __forceinline__ __bf16 f2bf(float f) {
  unsigned u = __builtin_bit_cast(unsigned, f);
  unsigned r = u + 0x7fffu + ((u >> 16) & 1u);   // round-to-nearest-even
  unsigned short h = (unsigned short)(r >> 16);
  return __builtin_bit_cast(__bf16, h);
}

// ---------------------------------------------------------------------------
// Async global -> LDS copy, 16 bytes per lane (CDNA5 ISA)
// VDST VGPR = LDS byte offset; VADDR = 64-bit global address (GV mode).
// Tracked by ASYNCcnt; completed with s_wait_asynccnt then barrier.
// ---------------------------------------------------------------------------
__device__ __forceinline__ void async_copy_b128(const __bf16* gsrc, __bf16* lds_dst) {
  unsigned lds = (unsigned)(unsigned long long)lds_dst;   // addr[31:0] = LDS offset
  unsigned long long ga = (unsigned long long)gsrc;
  asm volatile("global_load_async_to_lds_b128 %0, %1, off"
               :: "v"(lds), "v"(ga) : "memory");
}
__device__ __forceinline__ void wait_async_all() {
  asm volatile("s_wait_asynccnt 0x0" ::: "memory");
}

// ---------------------------------------------------------------------------
// WMMA fragment loaders (contiguous k order used by this code for both A and B, wave32)
//
// A (16x32 bf16, row-major source, stride lda):
//   lane l: row = l&15, k0 = (l>>4)*16
//   vector elems 0..7  = K k0..k0+7      (VGPR0..3)
//   vector elems 8..15 = K k0+8..k0+15   (VGPR4..7)
//
// B (32x16 bf16) loaded from a TRANSPOSED weight matrix WT[N][K] (stride ldk):
//   lane l: col n = l&15, kb = (l>>4)*16, elems e = K kb+e  -> 16 contiguous
// ---------------------------------------------------------------------------
__device__ __forceinline__ v16bf load_a_frag(const __bf16* base, int lda) {
  const int l = threadIdx.x & 31;
  const __bf16* p = base + (size_t)(l & 15) * lda + ((l >> 4) << 4);
  v8bf lo = *(const v8bf*)(p);
  v8bf hi = *(const v8bf*)(p + 8);
  v16bf a;
#pragma unroll
  for (int i = 0; i < 8; ++i) { a[i] = lo[i]; a[i + 8] = hi[i]; }
  return a;
}

__device__ __forceinline__ v16bf load_b_frag(const __bf16* __restrict__ baseT, int ldk) {
  const int l = threadIdx.x & 31;
  const __bf16* p = baseT + (size_t)(l & 15) * ldk + ((l >> 4) << 4);
  return *(const v16bf*)p;  // 32 contiguous bytes -> 2x global_load_b128
}

__device__ __forceinline__ v8f wmma_bf16(v16bf a, v16bf b, v8f c) {
  return __builtin_amdgcn_wmma_f32_16x16x32_bf16(
      /*neg_a=*/false, a, /*neg_b=*/false, b,
      /*c_mod=*/(short)0, c, /*reuse_a=*/false, /*reuse_b=*/false);
}

// ---------------------------------------------------------------------------
// Kernel: tiled transpose + f32->bf16 convert.  src[R][C] f32 -> dst[C][R] bf16
// R, C multiples of 32. block (32,8), grid (C/32, R/32)
// ---------------------------------------------------------------------------
__global__ void transpose_cvt_kernel(const float* __restrict__ src,
                                     __bf16* __restrict__ dst, int R, int C) {
  __shared__ float tile[32][33];
  const int c0 = blockIdx.x * 32, r0 = blockIdx.y * 32;
  const int tx = threadIdx.x, ty = threadIdx.y;
#pragma unroll
  for (int i = 0; i < 32; i += 8)
    tile[ty + i][tx] = src[(size_t)(r0 + ty + i) * C + c0 + tx];
  __syncthreads();
#pragma unroll
  for (int i = 0; i < 32; i += 8)
    dst[(size_t)(c0 + ty + i) * R + r0 + tx] = f2bf(tile[tx][ty + i]);
}

// ---------------------------------------------------------------------------
// Kernel: embedding gather + f32->bf16, into time-major x[T][B][E]
// grid (T, B), block 128; each thread converts 4 consecutive elements
// ---------------------------------------------------------------------------
__global__ void embed_kernel(const int* __restrict__ inputs,
                             const float* __restrict__ emb,
                             __bf16* __restrict__ x) {
  const int t = blockIdx.x, b = blockIdx.y;
  const int row = inputs[b * TLEN + t];
  const float4* src = (const float4*)(emb + (size_t)row * EMBED);
  __bf16* dst = x + ((size_t)t * BATCH + b) * EMBED;
  const int i = threadIdx.x;           // 128 threads * 4 = 512 = EMBED
  float4 v = src[i];
  v4bf o = { f2bf(v.x), f2bf(v.y), f2bf(v.z), f2bf(v.w) };
  *(v4bf*)(dst + 4 * i) = o;
}

// ---------------------------------------------------------------------------
// Kernel: one RNN layer, one timestep:
//   hout[16][1024] = tanh( xA[16][KX] @ Wx + hA[16][1024] @ Wh + bias )
// A panels (x_t and h_prev) are contiguous [16][K]; staged once per block
// into LDS via async copies, shared by all 8 waves.
// Weights pre-transposed: WxT[1024][KX], WhT[1024][1024].
// grid 8, block 256 (8 waves); one wave -> one 16x16 tile of hout.
// ---------------------------------------------------------------------------
template <int KX>
__global__ void rnn_step_kernel(const __bf16* __restrict__ xA,   // [16][KX] contiguous
                                const __bf16* __restrict__ WxT,
                                const __bf16* __restrict__ hA,   // [16][1024] contiguous
                                const __bf16* __restrict__ WhT,
                                const float* __restrict__ bias,
                                __bf16* __restrict__ hout) {
  __shared__ __align__(32) __bf16 sx[BATCH * KX];      // 16 or 32 KB
  __shared__ __align__(32) __bf16 sh[BATCH * HIDDEN];  // 32 KB
  const int tid = threadIdx.x;

  // ---- async stage A panels into LDS (16B per lane per issue) ----
#pragma unroll
  for (int r = 0; r < (BATCH * KX) / (8 * 256); ++r) {
    const int idx = (r * 256 + tid) * 8;
    async_copy_b128(xA + idx, sx + idx);
  }
#pragma unroll
  for (int r = 0; r < (BATCH * HIDDEN) / (8 * 256); ++r) {
    const int idx = (r * 256 + tid) * 8;
    async_copy_b128(hA + idx, sh + idx);
  }
  wait_async_all();
  __syncthreads();

  const int wave = tid >> 5;                          // wave32
  const int n0 = (blockIdx.x * 8 + wave) * 16;
  v8f c = {};
  const __bf16* BxT = WxT + (size_t)n0 * KX;
  const __bf16* BhT = WhT + (size_t)n0 * HIDDEN;
#pragma unroll 4
  for (int k = 0; k < KX; k += 32) {
    v16bf a = load_a_frag(sx + k, KX);                // ds_load_b128 x2
    v16bf b = load_b_frag(BxT + k, KX);               // global_load_b128 x2
    c = wmma_bf16(a, b, c);
  }
#pragma unroll 4
  for (int k = 0; k < HIDDEN; k += 32) {
    v16bf a = load_a_frag(sh + k, HIDDEN);
    v16bf b = load_b_frag(BhT + k, HIDDEN);
    c = wmma_bf16(a, b, c);
  }
  const int l = tid & 31;
  const int n = n0 + (l & 15);
  const float bv = bias[n];
  const int mb = (l >> 4) * 8;                        // C/D layout: M = j + 8*(l>>4)
#pragma unroll
  for (int j = 0; j < 8; ++j)
    hout[(size_t)(mb + j) * HIDDEN + n] = f2bf(tanhf(c[j] + bv));
}

// ---------------------------------------------------------------------------
// Kernel: FC GEMM  out[b][t][v] = hs[t][b][:] . fc_wT[v][:] + fc_b[v]
// hs: [T*B=4096][1024] bf16 (row m = t*16+b), fwT: [32000][1024] bf16.
// The 16x1024 A panel for this block's t is staged once into LDS (async),
// then shared by all 8 waves across the whole k loop; B streams from L2
// (64 MB bf16 fc_wT resides in the 192 MB L2 after first touch).
// grid (125, 256), block 256 (8 waves); wave -> two 16x16 N-tiles (32 cols).
// ---------------------------------------------------------------------------
__global__ void fc_gemm_kernel(const __bf16* __restrict__ hs,
                               const __bf16* __restrict__ fwT,
                               const float* __restrict__ fcb,
                               float* __restrict__ out) {
  __shared__ __align__(32) __bf16 As[BATCH * HIDDEN];  // 32 KB
  const int tid = threadIdx.x;
  const int t = blockIdx.y;
  const __bf16* A = hs + (size_t)t * BATCH * HIDDEN;

  // ---- async stage the A panel (16 x 1024 bf16 = 32 KB) ----
#pragma unroll
  for (int r = 0; r < (BATCH * HIDDEN) / (8 * 256); ++r) {  // 8 issues/thread
    const int idx = (r * 256 + tid) * 8;
    async_copy_b128(A + idx, As + idx);
  }
  wait_async_all();
  __syncthreads();

  const int wave = tid >> 5;
  const int n0 = blockIdx.x * 256 + wave * 32;
  const __bf16* B0 = fwT + (size_t)n0 * HIDDEN;
  const __bf16* B1 = fwT + (size_t)(n0 + 16) * HIDDEN;
  v8f c0 = {}, c1 = {};
#pragma unroll 4
  for (int k = 0; k < HIDDEN; k += 32) {
    __builtin_prefetch(B0 + k + 256, 0, 0);           // global_prefetch
    v16bf a  = load_a_frag(As + k, HIDDEN);           // ds_load_b128 x2
    v16bf b0 = load_b_frag(B0 + k, HIDDEN);           // global_load_b128 x2
    v16bf b1 = load_b_frag(B1 + k, HIDDEN);
    c0 = wmma_bf16(a, b0, c0);
    c1 = wmma_bf16(a, b1, c1);
  }
  const int l = tid & 31;
  const int mb = (l >> 4) * 8;
  const int nc = n0 + (l & 15);
  const float bv0 = fcb[nc], bv1 = fcb[nc + 16];
  const size_t TV = (size_t)TLEN * VOCAB;
  const size_t base = (size_t)t * VOCAB + nc;
#pragma unroll
  for (int j = 0; j < 8; ++j) {
    out[(size_t)(mb + j) * TV + base]      = c0[j] + bv0;
    out[(size_t)(mb + j) * TV + base + 16] = c1[j] + bv1;
  }
}

// ---------------------------------------------------------------------------
// Launcher
// ---------------------------------------------------------------------------
extern "C" void kernel_launch(void* const* d_in, const int* in_sizes, int n_in,
                              void* d_out, int out_size, void* d_ws, size_t ws_size,
                              hipStream_t stream) {
  (void)in_sizes; (void)n_in; (void)out_size; (void)ws_size;

  const int*   inputs = (const int*)  d_in[0];
  const float* emb    = (const float*)d_in[1];
  const float* W_xh0  = (const float*)d_in[2];
  const float* W_hh0  = (const float*)d_in[3];
  const float* b_h0   = (const float*)d_in[4];
  const float* W_xh1  = (const float*)d_in[5];
  const float* W_hh1  = (const float*)d_in[6];
  const float* b_h1   = (const float*)d_in[7];
  const float* fc_w   = (const float*)d_in[8];
  const float* fc_b   = (const float*)d_in[9];
  float* out = (float*)d_out;

  // ---- workspace carve-up (bytes) ----
  char* ws = (char*)d_ws;
  size_t off = 0;
  auto carve = [&](size_t bytes) { char* p = ws + off; off += (bytes + 255) & ~(size_t)255; return p; };
  __bf16* x      = (__bf16*)carve((size_t)TLEN * BATCH * EMBED * 2);        //  4 MB  [T][B][E]
  __bf16* WxT0   = (__bf16*)carve((size_t)HIDDEN * EMBED * 2);              //  1 MB  [1024][512]
  __bf16* WhT0   = (__bf16*)carve((size_t)HIDDEN * HIDDEN * 2);             //  2 MB
  __bf16* WxT1   = (__bf16*)carve((size_t)HIDDEN * HIDDEN * 2);             //  2 MB
  __bf16* WhT1   = (__bf16*)carve((size_t)HIDDEN * HIDDEN * 2);             //  2 MB
  __bf16* fwT    = (__bf16*)carve((size_t)VOCAB * HIDDEN * 2);              // 64 MB  [32000][1024]
  __bf16* h0_all = (__bf16*)carve((size_t)(TLEN + 1) * BATCH * HIDDEN * 2); // 8.4 MB
  __bf16* hs_all = (__bf16*)carve((size_t)(TLEN + 1) * BATCH * HIDDEN * 2); // 8.4 MB

  const size_t HSTEP = (size_t)BATCH * HIDDEN;   // elems per timestep slot

  // zero the t = -1 hidden states (slot 0)
  hipMemsetAsync(h0_all, 0, HSTEP * 2, stream);
  hipMemsetAsync(hs_all, 0, HSTEP * 2, stream);

  // ---- weight transpose + bf16 convert ----
  dim3 tb(32, 8);
  transpose_cvt_kernel<<<dim3(HIDDEN / 32, EMBED  / 32), tb, 0, stream>>>(W_xh0, WxT0, EMBED,  HIDDEN);
  transpose_cvt_kernel<<<dim3(HIDDEN / 32, HIDDEN / 32), tb, 0, stream>>>(W_hh0, WhT0, HIDDEN, HIDDEN);
  transpose_cvt_kernel<<<dim3(HIDDEN / 32, HIDDEN / 32), tb, 0, stream>>>(W_xh1, WxT1, HIDDEN, HIDDEN);
  transpose_cvt_kernel<<<dim3(HIDDEN / 32, HIDDEN / 32), tb, 0, stream>>>(W_hh1, WhT1, HIDDEN, HIDDEN);
  transpose_cvt_kernel<<<dim3(VOCAB  / 32, HIDDEN / 32), tb, 0, stream>>>(fc_w,  fwT,  HIDDEN, VOCAB);

  // ---- embedding gather (time-major, bf16) ----
  embed_kernel<<<dim3(TLEN, BATCH), 128, 0, stream>>>(inputs, emb, x);

  // ---- sequential 2-layer RNN scan (stream order = dependency order) ----
  for (int t = 0; t < TLEN; ++t) {
    rnn_step_kernel<EMBED><<<8, 256, 0, stream>>>(
        x + (size_t)t * BATCH * EMBED, WxT0,
        h0_all + (size_t)t * HSTEP, WhT0, b_h0,
        h0_all + (size_t)(t + 1) * HSTEP);
    rnn_step_kernel<HIDDEN><<<8, 256, 0, stream>>>(
        h0_all + (size_t)(t + 1) * HSTEP, WxT1,
        hs_all + (size_t)t * HSTEP, WhT1, b_h1,
        hs_all + (size_t)(t + 1) * HSTEP);
  }

  // ---- big FC GEMM over all stacked hidden states ----
  fc_gemm_kernel<<<dim3(VOCAB / 256, TLEN), 256, 0, stream>>>(
      hs_all + HSTEP /* slot 1 == t 0 */, fwT, fc_b, out);
}
